// AttentionPairBias_49744311222641
// MI455X (gfx1250) — compile-verified, matrix operand order checked by reading, not run on hardware
//
#include <hip/hip_runtime.h>
#include <hip/hip_bf16.h>

typedef __attribute__((ext_vector_type(16))) _Float16 v16h;
typedef __attribute__((ext_vector_type(8)))  float    v8f;

#define N_SEQ 768
#define DMODEL 512
#define NHEAD 16
#define DHEAD 32
#define CZ 128

// A/B f16 fragment K-offset for lane `l`, element `e` (16x16x32 WMMA):
//   elements 0..7  -> K = kbase + e        (kbase = (l>>4)*8)
//   elements 8..15 -> K = kbase + 8 + e
__device__ __forceinline__ int frag_ko(int lane, int e) {
    return ((lane >> 4) << 3) + (e < 8 ? e : e + 8);
}

// Load 8 consecutive f32 (16B-aligned) and pack into fragment halves e0..e0+7.
__device__ __forceinline__ void pack8(const float* __restrict__ p, v16h& f, int e0) {
    float4 a = *reinterpret_cast<const float4*>(p);
    float4 b = *reinterpret_cast<const float4*>(p + 4);
    f[e0 + 0] = (_Float16)a.x; f[e0 + 1] = (_Float16)a.y;
    f[e0 + 2] = (_Float16)a.z; f[e0 + 3] = (_Float16)a.w;
    f[e0 + 4] = (_Float16)b.x; f[e0 + 5] = (_Float16)b.y;
    f[e0 + 6] = (_Float16)b.z; f[e0 + 7] = (_Float16)b.w;
}

// ---------------------------------------------------------------------------
// Kernel 1: fused QKVG projection.  qkvg[n][0:512]=q+bq, [512:1024]=k,
// [1024:1536]=v, [1536:2048]=g.   768x2048 output, K=512, wave -> 16x16 tile.
// ---------------------------------------------------------------------------
__global__ __launch_bounds__(256) void qkvg_proj_kernel(
    const float* __restrict__ s, const float* __restrict__ Wq,
    const float* __restrict__ bq, const float* __restrict__ Wk,
    const float* __restrict__ Wv, const float* __restrict__ Wg,
    float* __restrict__ qkvg) {
    const int lane = threadIdx.x & 31;
    const int warp = threadIdx.x >> 5;
    const int wid  = blockIdx.x * 8 + warp;       // 48 * 128 = 6144 tiles
    const int tm   = wid >> 7;                    // 0..47  (query rows / 16)
    const int tj   = wid & 127;                   // 0..127 (out cols / 16)
    const int which = tj >> 5;                    // 0=q 1=k 2=v 3=g
    const float* W = (which == 0) ? Wq : (which == 1) ? Wk
                   : (which == 2) ? Wv : Wg;
    const int ln = lane & 15;
    const int kb = (lane >> 4) << 3;
    const int arow = tm * 16 + ln;
    const int jcol = ((tj & 31) << 4) + ln;       // feature row inside W

    v8f acc = {};
    for (int kk = 0; kk < DMODEL; kk += 32) {
        v16h a, b;
        pack8(s + arow * DMODEL + kk + kb,      a, 0);
        pack8(s + arow * DMODEL + kk + kb + 16, a, 8);
        pack8(W + jcol * DMODEL + kk + kb,      b, 0);
        pack8(W + jcol * DMODEL + kk + kb + 16, b, 8);
        acc = __builtin_amdgcn_wmma_f32_16x16x32_f16(
                  false, a, false, b, (short)0, acc, false, false);
    }
    const float bias = (which == 0) ? bq[jcol] : 0.f;
    const int col = tj * 16 + ln;
    const int hi  = lane >> 4;
#pragma unroll
    for (int r = 0; r < 8; ++r) {
        int m = hi * 8 + r;
        qkvg[(size_t)(tm * 16 + m) * 2048 + col] = acc[r] + bias;
    }
}

// ---------------------------------------------------------------------------
// Kernel 2: pair bias (HBM-bound: streams z once, 302 MB).
// Each wave processes PB_TPW contiguous 16-row tiles of z (8 KB each,
// contiguous in memory).  Tiles are copied global->LDS with
// global_load_async_to_lds_b128 (ASYNCcnt), double-buffered so tile t+1
// streams in while tile t is LayerNorm'd + WMMA'd against Wz^T.
// bias layout: [q][k][h]  (coalesced, non-temporal stores).
// ---------------------------------------------------------------------------
#define PB_TPW 8
__global__ __launch_bounds__(128) void pairbias_kernel(
    const float* __restrict__ z, const float* __restrict__ ln_w,
    const float* __restrict__ ln_b, const float* __restrict__ Wz,
    float* __restrict__ bias) {
    __shared__ float buf[4][2][16][CZ];           // 64 KB: 2 x 8KB per wave
    const int lane = threadIdx.x & 31;
    const int warp = threadIdx.x >> 5;
    const int t0   = (blockIdx.x * 4 + warp) * PB_TPW;   // 36864 tiles total
    const int row  = lane & 15;                   // pair row within tile
    const int half = lane >> 4;                   // 64-elem half of z row
    const int hsel = lane & 15;                   // head column for B
    const int hi   = lane >> 4;

    // per-lane LDS byte addresses for the async tile copies
    uint32_t lds_lane[2];
    lds_lane[0] = (uint32_t)(uintptr_t)(&buf[warp][0][0][0]) + lane * 16;
    lds_lane[1] = (uint32_t)(uintptr_t)(&buf[warp][1][0][0]) + lane * 16;

    // prefetch tile 0 into buffer 0 (16 x 512B coalesced async copies)
    {
        const char* g = (const char*)(z + (size_t)t0 * 16 * CZ) + lane * 16;
#pragma unroll
        for (int i = 0; i < 16; ++i)
            asm volatile("global_load_async_to_lds_b128 %0, %1, off offset:%2"
                         :: "v"(lds_lane[0]), "v"(g), "i"(i * 512) : "memory");
    }

    for (int tt = 0; tt < PB_TPW; ++tt) {
        const int t    = t0 + tt;
        const int bsel = tt & 1;
        if (tt + 1 < PB_TPW) {
            // stream next tile into the other buffer, then wait for current
            const char* g =
                (const char*)(z + (size_t)(t + 1) * 16 * CZ) + lane * 16;
#pragma unroll
            for (int i = 0; i < 16; ++i)
                asm volatile(
                    "global_load_async_to_lds_b128 %0, %1, off offset:%2"
                    :: "v"(lds_lane[bsel ^ 1]), "v"(g), "i"(i * 512)
                    : "memory");
            asm volatile("s_wait_asynccnt 0x10" ::: "memory");
        } else {
            asm volatile("s_wait_asynccnt 0x0" ::: "memory");
        }

        // LayerNorm stats: lane pair (row, row+16) covers one 128-elem row
        float sum = 0.f, sq = 0.f;
#pragma unroll
        for (int i = 0; i < 64; i += 4) {
            float4 x = *reinterpret_cast<const float4*>(
                &buf[warp][bsel][row][half * 64 + i]);
            sum += x.x + x.y + x.z + x.w;
            sq  += x.x * x.x + x.y * x.y + x.z * x.z + x.w * x.w;
        }
        sum += __shfl_xor(sum, 16);
        sq  += __shfl_xor(sq, 16);
        const float mean = sum * (1.f / CZ);
        const float var  = sq * (1.f / CZ) - mean * mean;
        const float rstd = rsqrtf(var + 1e-5f);

        v8f acc = {};
#pragma unroll
        for (int c = 0; c < 4; ++c) {
            v16h a, b;
#pragma unroll
            for (int e = 0; e < 16; ++e) {
                int k = c * 32 + frag_ko(lane, e);
                float x = buf[warp][bsel][row][k];
                a[e] = (_Float16)((x - mean) * rstd * ln_w[k] + ln_b[k]);
                b[e] = (_Float16)Wz[hsel * CZ + k];
            }
            acc = __builtin_amdgcn_wmma_f32_16x16x32_f16(
                      false, a, false, b, (short)0, acc, false, false);
        }
        const int q   = t / 48;
        const int kbt = t % 48;
#pragma unroll
        for (int r = 0; r < 8; ++r) {
            int ki = kbt * 16 + hi * 8 + r;
            __builtin_nontemporal_store(
                acc[r], &bias[((size_t)(q * N_SEQ + ki)) * NHEAD + hsel]);
        }
    }
}

// ---------------------------------------------------------------------------
// Kernel 3: flash attention.  Block = 16 waves (one head per wave) x one
// 16-query tile.  Online softmax, 32-key steps: 2 QK wmma + 2 PV wmma.
// P (D-layout) transposed to A-layout through per-wave LDS.
// ---------------------------------------------------------------------------
__global__ __launch_bounds__(512) void attn_kernel(
    const float* __restrict__ qkvg, const float* __restrict__ bias,
    float* __restrict__ o) {
    __shared__ _Float16 ldsP[NHEAD][16][34];      // padded rows, ~17 KB
    const int lane = threadIdx.x & 31;
    const int h    = threadIdx.x >> 5;            // head = wave id
    const int q0   = blockIdx.x * 16;
    const int ln   = lane & 15;
    const int hi   = lane >> 4;
    const int kb   = hi << 3;
    const float scale = 0.1767766952966369f;      // 32^-0.5

    v16h qa;                                      // Q fragment, pre-scaled
    {
        const float* qp = qkvg + (size_t)(q0 + ln) * 2048 + h * DHEAD;
        pack8(qp + kb,      qa, 0);
        pack8(qp + kb + 16, qa, 8);
#pragma unroll
        for (int e = 0; e < 16; ++e)
            qa[e] = (_Float16)((float)qa[e] * scale);
    }
    float M[8], L[8];
    v8f O0 = {}, O1 = {};
#pragma unroll
    for (int r = 0; r < 8; ++r) { M[r] = -1e30f; L[r] = 0.f; }

    for (int k0 = 0; k0 < N_SEQ; k0 += 32) {
        v16h bk0, bk1;
        const float* kp0 = qkvg + (size_t)(k0 + ln) * 2048 + 512 + h * DHEAD;
        const float* kp1 = kp0 + 16 * 2048;
        pack8(kp0 + kb,      bk0, 0);
        pack8(kp0 + kb + 16, bk0, 8);
        pack8(kp1 + kb,      bk1, 0);
        pack8(kp1 + kb + 16, bk1, 8);
        v8f zero = {};
        v8f S0 = __builtin_amdgcn_wmma_f32_16x16x32_f16(
                     false, qa, false, bk0, (short)0, zero, false, false);
        v8f S1 = __builtin_amdgcn_wmma_f32_16x16x32_f16(
                     false, qa, false, bk1, (short)0, zero, false, false);
        float nm[8];
#pragma unroll
        for (int r = 0; r < 8; ++r) {
            int m = hi * 8 + r;
            const float* bp =
                bias + ((size_t)((q0 + m) * N_SEQ + k0 + ln)) * NHEAD + h;
            S0[r] += __builtin_nontemporal_load(bp);
            S1[r] += __builtin_nontemporal_load(bp + 16 * NHEAD);
            nm[r] = fmaxf(S0[r], S1[r]);
        }
#pragma unroll
        for (int off = 8; off >= 1; off >>= 1)
#pragma unroll
            for (int r = 0; r < 8; ++r)
                nm[r] = fmaxf(nm[r], __shfl_xor(nm[r], off));
        float rs[8];
#pragma unroll
        for (int r = 0; r < 8; ++r) {
            float newM  = fmaxf(M[r], nm[r]);
            float alpha = __expf(M[r] - newM);
            S0[r] = __expf(S0[r] - newM);
            S1[r] = __expf(S1[r] - newM);
            rs[r] = S0[r] + S1[r];
            O0[r] *= alpha; O1[r] *= alpha;
            L[r] *= alpha;
            M[r] = newM;
        }
#pragma unroll
        for (int off = 8; off >= 1; off >>= 1)
#pragma unroll
            for (int r = 0; r < 8; ++r)
                rs[r] += __shfl_xor(rs[r], off);
#pragma unroll
        for (int r = 0; r < 8; ++r) L[r] += rs[r];

        // transpose P: D-layout -> A-layout (f16) via LDS
#pragma unroll
        for (int r = 0; r < 8; ++r) {
            int m = hi * 8 + r;
            ldsP[h][m][ln]      = (_Float16)S0[r];
            ldsP[h][m][16 + ln] = (_Float16)S1[r];
        }
        __syncthreads();
        v16h pa;
#pragma unroll
        for (int e = 0; e < 16; ++e)
            pa[e] = ldsP[h][ln][frag_ko(lane, e)];

        v16h bv0, bv1;
#pragma unroll
        for (int e = 0; e < 16; ++e) {
            int kl = frag_ko(lane, e);
            const float* vp =
                qkvg + (size_t)(k0 + kl) * 2048 + 1024 + h * DHEAD;
            bv0[e] = (_Float16)vp[ln];
            bv1[e] = (_Float16)vp[16 + ln];
        }
        O0 = __builtin_amdgcn_wmma_f32_16x16x32_f16(
                 false, pa, false, bv0, (short)0, O0, false, false);
        O1 = __builtin_amdgcn_wmma_f32_16x16x32_f16(
                 false, pa, false, bv1, (short)0, O1, false, false);
        __syncthreads();
    }
#pragma unroll
    for (int r = 0; r < 8; ++r) {
        int m = hi * 8 + r;
        float inv = 1.f / L[r];
        o[(size_t)(q0 + m) * DMODEL + h * DHEAD + ln]      = O0[r] * inv;
        o[(size_t)(q0 + m) * DMODEL + h * DHEAD + 16 + ln] = O1[r] * inv;
    }
}

// ---------------------------------------------------------------------------
// Kernel 4: out = (o * sigmoid(g)) @ Wo^T.   768x512, K=512.
// ---------------------------------------------------------------------------
__global__ __launch_bounds__(256) void outproj_kernel(
    const float* __restrict__ o, const float* __restrict__ qkvg,
    const float* __restrict__ Wo, float* __restrict__ out) {
    const int lane = threadIdx.x & 31;
    const int warp = threadIdx.x >> 5;
    const int wid  = blockIdx.x * 8 + warp;       // 48 * 32 = 1536 tiles
    const int tm   = wid >> 5;
    const int tj   = wid & 31;
    const int ln   = lane & 15;
    const int kb   = (lane >> 4) << 3;
    const int row  = tm * 16 + ln;
    const int jcol = tj * 16 + ln;

    v8f acc = {};
    for (int kk = 0; kk < DMODEL; kk += 32) {
        v16h a, b;
#pragma unroll
        for (int run = 0; run < 2; ++run) {
            int k0 = kk + kb + run * 16;
            const float* op = o    + (size_t)row * DMODEL + k0;
            const float* gp = qkvg + (size_t)row * 2048 + 1536 + k0;
            int e0 = run * 8;
#pragma unroll
            for (int i = 0; i < 8; ++i)
                a[e0 + i] = (_Float16)(op[i] / (1.f + __expf(-gp[i])));
            pack8(Wo + (size_t)jcol * DMODEL + k0, b, e0);
        }
        acc = __builtin_amdgcn_wmma_f32_16x16x32_f16(
                  false, a, false, b, (short)0, acc, false, false);
    }
    const int hi = lane >> 4;
#pragma unroll
    for (int r = 0; r < 8; ++r) {
        int m = hi * 8 + r;
        out[(size_t)(tm * 16 + m) * DMODEL + tj * 16 + ln] = acc[r];
    }
}

extern "C" void kernel_launch(void* const* d_in, const int* in_sizes, int n_in,
                              void* d_out, int out_size, void* d_ws,
                              size_t ws_size, hipStream_t stream) {
    (void)in_sizes; (void)n_in; (void)out_size; (void)ws_size;
    const float* s    = (const float*)d_in[0];
    const float* z    = (const float*)d_in[1];
    const float* Wq   = (const float*)d_in[2];
    const float* bq   = (const float*)d_in[3];
    const float* Wk   = (const float*)d_in[4];
    const float* Wv   = (const float*)d_in[5];
    const float* Wg   = (const float*)d_in[6];
    const float* Wo   = (const float*)d_in[7];
    const float* ln_w = (const float*)d_in[8];
    const float* ln_b = (const float*)d_in[9];
    const float* Wz   = (const float*)d_in[10];

    float* ws    = (float*)d_ws;
    float* qkvg  = ws;                                    // 768*2048 f32
    float* biasb = qkvg + (size_t)N_SEQ * 2048;           // 768*768*16 f32
    float* obuf  = biasb + (size_t)N_SEQ * N_SEQ * NHEAD; // 768*512 f32
    float* out   = (float*)d_out;

    hipLaunchKernelGGL(qkvg_proj_kernel, dim3(768), dim3(256), 0, stream,
                       s, Wq, bq, Wk, Wv, Wg, qkvg);
    hipLaunchKernelGGL(pairbias_kernel, dim3(1152), dim3(128), 0, stream,
                       z, ln_w, ln_b, Wz, biasb);
    hipLaunchKernelGGL(attn_kernel, dim3(48), dim3(512), 0, stream,
                       qkvg, biasb, obuf);
    hipLaunchKernelGGL(outproj_kernel, dim3(192), dim3(256), 0, stream,
                       obuf, qkvg, Wo, out);
}
